// MultiHeadAttention_54185307406780
// MI455X (gfx1250) — matrix operand order repaired and checked by reading, not compiled, not run on hardware
//
#include <hip/hip_runtime.h>

typedef __bf16 bf16_t;
typedef __attribute__((ext_vector_type(16))) __bf16 v16bf;
typedef __attribute__((ext_vector_type(4)))  __bf16 v4bf;
typedef __attribute__((ext_vector_type(8)))  float  v8f;

#define L_Q    2048
#define L_K    2048
#define BATCH  2
#define DMODEL 1024
#define NHEADS 16
#define DK     64

// LDS B-tile pitch: 40 bf16 (80B = 20 banks). gcd(20,64)=4 -> 16 distinct
// bank groups across the 16 rows a half-wave reads: conflict-free b128 reads.
#define BPITCH 40
#define BUFELEMS (64 * BPITCH)

// ---------------- CDNA5 async copy (ASYNCcnt-tracked DMA to LDS) -----------
static __device__ __forceinline__ unsigned lds_addr_of(const void* p) {
  // Flat LDS addresses carry the LDS byte offset in the low 32 bits.
  return (unsigned)(uintptr_t)p;
}
static __device__ __forceinline__ void async_copy_b128(unsigned lds_addr,
                                                       const void* gptr) {
  asm volatile("global_load_async_to_lds_b128 %0, %1, off"
               :: "v"(lds_addr), "v"((unsigned long long)(uintptr_t)gptr)
               : "memory");
}
static __device__ __forceinline__ void wait_async0() {
  asm volatile("s_wait_asynccnt 0x0" ::: "memory");
}

// ---------------- WMMA fragment helpers (layouts per CDNA5 ISA 7.12.2) -----

// A-matrix 16x32 bf16: lane L holds row M=L&15; lanes 0-15 K={0..7,16..23},
// lanes 16-31 K={8..15,24..31}. Two 16-byte chunks per lane.
static __device__ __forceinline__ v16bf load_a_frag(const bf16_t* __restrict__ row,
                                                    int k0, int hl) {
  v16bf f;
  uint4* d = reinterpret_cast<uint4*>(&f);
  d[0] = *reinterpret_cast<const uint4*>(row + k0 + hl * 8);
  d[1] = *reinterpret_cast<const uint4*>(row + k0 + 16 + hl * 8);
  return f;
}

// B-matrix 32x16 bf16 (N-major source): lane L holds column N=L&15;
// lanes 0-15 K={0..7,16..23}, lanes 16-31 K={8..15,24..31}. Two 16B chunks.
static __device__ __forceinline__ v16bf load_b_frag(const bf16_t* __restrict__ row,
                                                    int k0, int hl) {
  v16bf f;
  uint4* d = reinterpret_cast<uint4*>(&f);
  const bf16_t* p = row + k0 + hl * 8;
  d[0] = *reinterpret_cast<const uint4*>(p);
  d[1] = *reinterpret_cast<const uint4*>(p + 16);
  return f;
}

// Same B fragment sourced from an LDS row (row already K-sliced, BPITCH pitch).
static __device__ __forceinline__ v16bf load_b_frag_lds(const bf16_t* row, int hl) {
  v16bf f;
  uint4* d = reinterpret_cast<uint4*>(&f);
  const uint4* p = reinterpret_cast<const uint4*>(row);
  d[0] = p[hl];
  d[1] = p[2 + hl];
  return f;
}

// A-fragment built from fp32 source (attention probabilities), cvt inline.
static __device__ __forceinline__ v16bf load_a_frag_f32(const float* __restrict__ row,
                                                        int k0, int hl) {
  v16bf f;
  const float* p0 = row + k0 + hl * 8;
  const float* p1 = row + k0 + 16 + hl * 8;
#pragma unroll
  for (int i = 0; i < 8; ++i) {
    f[i]     = (bf16_t)p0[i];
    f[8 + i] = (bf16_t)p1[i];
  }
  return f;
}

static __device__ __forceinline__ v8f wmma_bf16(v16bf a, v16bf b, v8f c) {
  return __builtin_amdgcn_wmma_f32_16x16x32_bf16(false, a, false, b,
                                                 (short)0, c, false, false);
}

// ---------------- fp32 -> bf16 bulk convert ---------------------------------
__global__ __launch_bounds__(256) void cvt_f32_bf16(const float* __restrict__ in,
                                                    bf16_t* __restrict__ out, int n4) {
  int i = blockIdx.x * blockDim.x + threadIdx.x;
  if (i >= n4) return;
  float4 f = reinterpret_cast<const float4*>(in)[i];
  v4bf o;
  o[0] = (bf16_t)f.x; o[1] = (bf16_t)f.y; o[2] = (bf16_t)f.z; o[3] = (bf16_t)f.w;
  reinterpret_cast<v4bf*>(out)[i] = o;
}

// ---------------- Projection: Y = X @ W^T + b, scatter into head layout -----
// X: (L, BATCH, DMODEL) bf16; W: (DMODEL, DMODEL) bf16 row-major (N x K).
// B tile (64 x 32 per K-step) staged via async DMA to LDS, double-buffered
// with a ping-pong offset (loop kept rolled so accumulators stay in place).
// mode 0: out (BATCH, H, L, DK)    mode 1: out (BATCH, H, DK, L) (transposed V)
__global__ __launch_bounds__(256) void proj_kernel(const bf16_t* __restrict__ X,
                                                   const bf16_t* __restrict__ W,
                                                   const float* __restrict__ bias,
                                                   bf16_t* __restrict__ out,
                                                   int transpose_out) {
  __shared__ __align__(16) bf16_t sB[2 * BUFELEMS];

  const int lane = threadIdx.x & 31;
  const int hl   = lane >> 4;
  const int r    = lane & 15;
  const int wave = threadIdx.x >> 5;
  const int batch = blockIdx.z;
  const int m0 = blockIdx.x * 128 + wave * 16;
  const int n0 = blockIdx.y * 64;

  // Per-thread async-copy assignment: 256 threads x 16B = 4KB = 64 rows x 32.
  const int tid    = threadIdx.x;
  const int crow   = tid >> 2;          // 0..63  (B tile row)
  const int celem  = (tid & 3) * 8;     // element offset 0,8,16,24
  const bf16_t* Wcopy = W + (size_t)(n0 + crow) * DMODEL + celem;
  const unsigned copyoff = (unsigned)(crow * BPITCH + celem);

  const bf16_t* Arow = X + ((size_t)(m0 + r) * BATCH + batch) * DMODEL;

  v8f zero = {0.f, 0.f, 0.f, 0.f, 0.f, 0.f, 0.f, 0.f};
  v8f acc[4] = {zero, zero, zero, zero};

  const int NSTEP = DMODEL / 32;
  unsigned curoff = 0, nxtoff = BUFELEMS;
  async_copy_b128(lds_addr_of(&sB[copyoff]), Wcopy);     // prime buffer 0

#pragma unroll 1
  for (int step = 0; step < NSTEP; ++step) {
    const int k0 = step * 32;
    wait_async0();
    __syncthreads();                                     // all waves' DMA landed
    if (step + 1 < NSTEP)
      async_copy_b128(lds_addr_of(&sB[nxtoff + copyoff]), Wcopy + k0 + 32);
    __builtin_prefetch(Arow + k0 + 64, 0, 1);
    v16bf a = load_a_frag(Arow, k0, hl);
    const bf16_t* bbase = &sB[curoff];
#pragma unroll
    for (int t = 0; t < 4; ++t) {
      v16bf b = load_b_frag_lds(bbase + (t * 16 + r) * BPITCH, hl);
      acc[t] = wmma_bf16(a, b, acc[t]);
    }
    unsigned tmp = curoff; curoff = nxtoff; nxtoff = tmp;
  }

  const int cn = lane & 15;
#pragma unroll
  for (int t = 0; t < 4; ++t) {
    const int n  = n0 + t * 16 + cn;
    const int h  = n >> 6;
    const int dk = n & 63;
    const float bv = bias[n];
#pragma unroll
    for (int rr = 0; rr < 8; ++rr) {
      const int m = m0 + hl * 8 + rr;   // sequence index l
      float v = acc[t][rr] + bv;
      size_t idx;
      if (!transpose_out)
        idx = (((size_t)(batch * NHEADS + h) * L_Q) + m) * DK + dk;
      else
        idx = (((size_t)(batch * NHEADS + h) * DK) + dk) * L_K + m;
      out[idx] = (bf16_t)v;
    }
  }
}

// ---------------- Scores: S = Q K^T / sqrt(dk), masked -> fp32 --------------
// K = 64 (two WMMA steps): direct loads, no staging needed.
__global__ __launch_bounds__(256) void scores_kernel(const bf16_t* __restrict__ Qb,
                                                     const bf16_t* __restrict__ Kb,
                                                     const int* __restrict__ mask,
                                                     float* __restrict__ attn) {
  const int lane = threadIdx.x & 31;
  const int hl   = lane >> 4;
  const int r    = lane & 15;
  const int wave = threadIdx.x >> 5;
  const int z  = blockIdx.z;                 // b*NHEADS + h
  const int m0 = blockIdx.x * 128 + wave * 16;
  const int n0 = blockIdx.y * 64;

  const bf16_t* Qh = Qb + (size_t)z * L_Q * DK;
  const bf16_t* Kh = Kb + (size_t)z * L_K * DK;
  const bf16_t* Arow = Qh + (size_t)(m0 + r) * DK;

  v8f zero = {0.f, 0.f, 0.f, 0.f, 0.f, 0.f, 0.f, 0.f};
  v8f acc[4] = {zero, zero, zero, zero};

#pragma unroll
  for (int k0 = 0; k0 < DK; k0 += 32) {
    v16bf a = load_a_frag(Arow, k0, hl);
#pragma unroll
    for (int t = 0; t < 4; ++t) {
      v16bf b = load_b_frag(Kh + (size_t)(n0 + t * 16 + r) * DK, k0, hl);
      acc[t] = wmma_bf16(a, b, acc[t]);
    }
  }

  const int cn = lane & 15;
  const float inv_scale = 0.125f;            // 1/sqrt(64)
#pragma unroll
  for (int t = 0; t < 4; ++t) {
    const int lk = n0 + t * 16 + cn;
#pragma unroll
    for (int rr = 0; rr < 8; ++rr) {
      const int lq = m0 + hl * 8 + rr;
      const int mv = mask[(size_t)lq * L_K + lk];
      float v = (mv == 0) ? -1.0e9f : acc[t][rr] * inv_scale;
      attn[((size_t)z * L_Q + lq) * L_K + lk] = v;
    }
  }
}

// ---------------- Row softmax, in place on fp32 attn ------------------------
__global__ __launch_bounds__(256) void softmax_kernel(float* __restrict__ attn) {
  __shared__ float red[256];
  const size_t row = blockIdx.x;
  float* p = attn + row * L_K;
  const int tid = threadIdx.x;

  float v[8];
  float mx = -3.0e38f;
#pragma unroll
  for (int i = 0; i < 8; ++i) {
    v[i] = p[tid + i * 256];
    mx = fmaxf(mx, v[i]);
  }
  red[tid] = mx; __syncthreads();
  for (int s = 128; s > 0; s >>= 1) {
    if (tid < s) red[tid] = fmaxf(red[tid], red[tid + s]);
    __syncthreads();
  }
  const float rmax = red[0]; __syncthreads();

  float sum = 0.f;
#pragma unroll
  for (int i = 0; i < 8; ++i) {
    v[i] = __expf(v[i] - rmax);
    sum += v[i];
  }
  red[tid] = sum; __syncthreads();
  for (int s = 128; s > 0; s >>= 1) {
    if (tid < s) red[tid] += red[tid + s];
    __syncthreads();
  }
  const float inv = 1.0f / red[0];
#pragma unroll
  for (int i = 0; i < 8; ++i) p[tid + i * 256] = v[i] * inv;
}

// ---------------- O = attn @ V, per (b,h); A cvt'd fp32->bf16 inline --------
// Vt: (BATCH, H, DK, L_K) bf16; B tile async-staged to LDS, double-buffered.
// Omid: (L_Q, BATCH, DMODEL) bf16
__global__ __launch_bounds__(256) void av_kernel(const float* __restrict__ attn,
                                                 const bf16_t* __restrict__ Vt,
                                                 bf16_t* __restrict__ Omid) {
  __shared__ __align__(16) bf16_t sB[2 * BUFELEMS];

  const int lane = threadIdx.x & 31;
  const int hl   = lane >> 4;
  const int r    = lane & 15;
  const int wave = threadIdx.x >> 5;
  const int z  = blockIdx.z;
  const int bb = z >> 4;
  const int h  = z & 15;
  const int m0 = blockIdx.x * 128 + wave * 16;

  const float*  Ah = attn + (size_t)z * L_Q * L_K;
  const bf16_t* Vh = Vt + (size_t)z * DK * L_K;
  const float*  Arow = Ah + (size_t)(m0 + r) * L_K;

  const int tid   = threadIdx.x;
  const int crow  = tid >> 2;           // 0..63  (dk row of V^T)
  const int celem = (tid & 3) * 8;
  const bf16_t* Vcopy = Vh + (size_t)crow * L_K + celem;
  const unsigned copyoff = (unsigned)(crow * BPITCH + celem);

  v8f zero = {0.f, 0.f, 0.f, 0.f, 0.f, 0.f, 0.f, 0.f};
  v8f acc[4] = {zero, zero, zero, zero};

  const int NSTEP = L_K / 32;
  unsigned curoff = 0, nxtoff = BUFELEMS;
  async_copy_b128(lds_addr_of(&sB[copyoff]), Vcopy);

#pragma unroll 1
  for (int step = 0; step < NSTEP; ++step) {
    const int k0 = step * 32;
    wait_async0();
    __syncthreads();
    if (step + 1 < NSTEP)
      async_copy_b128(lds_addr_of(&sB[nxtoff + copyoff]), Vcopy + k0 + 32);
    __builtin_prefetch(Arow + k0 + 64, 0, 1);
    v16bf a = load_a_frag_f32(Arow, k0, hl);
    const bf16_t* bbase = &sB[curoff];
#pragma unroll
    for (int t = 0; t < 4; ++t) {
      v16bf b = load_b_frag_lds(bbase + (t * 16 + r) * BPITCH, hl);
      acc[t] = wmma_bf16(a, b, acc[t]);
    }
    unsigned tmp = curoff; curoff = nxtoff; nxtoff = tmp;
  }

  const int cn = lane & 15;
#pragma unroll
  for (int t = 0; t < 4; ++t) {
    const int n = t * 16 + cn;           // dk
#pragma unroll
    for (int rr = 0; rr < 8; ++rr) {
      const int l = m0 + hl * 8 + rr;
      Omid[((size_t)l * BATCH + bb) * DMODEL + h * DK + n] = (bf16_t)acc[t][rr];
    }
  }
}

// ---------------- Final projection: Y = Omid @ Wo^T + b_o -> fp32 d_out -----
__global__ __launch_bounds__(256) void outproj_kernel(const bf16_t* __restrict__ Omid,
                                                      const bf16_t* __restrict__ Wo,
                                                      const float* __restrict__ bias,
                                                      float* __restrict__ Y) {
  __shared__ __align__(16) bf16_t sB[2 * BUFELEMS];

  const int lane = threadIdx.x & 31;
  const int hl   = lane >> 4;
  const int r    = lane & 15;
  const int wave = threadIdx.x >> 5;
  const int batch = blockIdx.z;
  const int m0 = blockIdx.x * 128 + wave * 16;
  const int n0 = blockIdx.y * 64;

  const bf16_t* Arow = Omid + ((size_t)(m0 + r) * BATCH + batch) * DMODEL;

  const int tid   = threadIdx.x;
  const int crow  = tid >> 2;
  const int celem = (tid & 3) * 8;
  const bf16_t* Wcopy = Wo + (size_t)(n0 + crow) * DMODEL + celem;
  const unsigned copyoff = (unsigned)(crow * BPITCH + celem);

  v8f zero = {0.f, 0.f, 0.f, 0.f, 0.f, 0.f, 0.f, 0.f};
  v8f acc[4] = {zero, zero, zero, zero};

  const int NSTEP = DMODEL / 32;
  unsigned curoff = 0, nxtoff = BUFELEMS;
  async_copy_b128(lds_addr_of(&sB[copyoff]), Wcopy);

#pragma unroll 1
  for (int step = 0; step < NSTEP; ++step) {
    const int k0 = step * 32;
    wait_async0();
    __syncthreads();
    if (step + 1 < NSTEP)
      async_copy_b128(lds_addr_of(&sB[nxtoff + copyoff]), Wcopy + k0 + 32);
    __builtin_prefetch(Arow + k0 + 64, 0, 1);
    v16bf a = load_a_frag(Arow, k0, hl);
    const bf16_t* bbase = &sB[curoff];
#pragma unroll
    for (int t = 0; t < 4; ++t) {
      v16bf b = load_b_frag_lds(bbase + (t * 16 + r) * BPITCH, hl);
      acc[t] = wmma_bf16(a, b, acc[t]);
    }
    unsigned tmp = curoff; curoff = nxtoff; nxtoff = tmp;
  }

  const int cn = lane & 15;
#pragma unroll
  for (int t = 0; t < 4; ++t) {
    const int n = n0 + t * 16 + cn;
    const float bv = bias[n];
#pragma unroll
    for (int rr = 0; rr < 8; ++rr) {
      const int m = m0 + hl * 8 + rr;
      Y[((size_t)m * BATCH + batch) * DMODEL + n] = acc[t][rr] + bv;
    }
  }
}

// ---------------- Host-side orchestration ----------------------------------
extern "C" void kernel_launch(void* const* d_in, const int* in_sizes, int n_in,
                              void* d_out, int out_size, void* d_ws, size_t ws_size,
                              hipStream_t stream) {
  const float* q    = (const float*)d_in[0];
  const float* k    = (const float*)d_in[1];
  const float* v    = (const float*)d_in[2];
  const int*   mask = (const int*)  d_in[3];
  const float* w_q  = (const float*)d_in[4];
  const float* b_q  = (const float*)d_in[5];
  const float* w_k  = (const float*)d_in[6];
  const float* b_k  = (const float*)d_in[7];
  const float* w_v  = (const float*)d_in[8];
  const float* b_v  = (const float*)d_in[9];
  const float* w_o  = (const float*)d_in[10];
  const float* b_o  = (const float*)d_in[11];

  const size_t X_ELEMS = (size_t)L_Q * BATCH * DMODEL;      // 4 Mi
  const size_t W_ELEMS = (size_t)DMODEL * DMODEL;           // 1 Mi
  const size_t H_ELEMS = (size_t)BATCH * NHEADS * L_Q * DK; // 4 Mi

  char* ws = (char*)d_ws;
  bf16_t* xq  = (bf16_t*)ws;  ws += X_ELEMS * sizeof(bf16_t);
  bf16_t* xk  = (bf16_t*)ws;  ws += X_ELEMS * sizeof(bf16_t);
  bf16_t* xv  = (bf16_t*)ws;  ws += X_ELEMS * sizeof(bf16_t);
  bf16_t* wqb = (bf16_t*)ws;  ws += W_ELEMS * sizeof(bf16_t);
  bf16_t* wkb = (bf16_t*)ws;  ws += W_ELEMS * sizeof(bf16_t);
  bf16_t* wvb = (bf16_t*)ws;  ws += W_ELEMS * sizeof(bf16_t);
  bf16_t* wob = (bf16_t*)ws;  ws += W_ELEMS * sizeof(bf16_t);
  bf16_t* Qb  = (bf16_t*)ws;  ws += H_ELEMS * sizeof(bf16_t);
  bf16_t* Kb  = (bf16_t*)ws;  ws += H_ELEMS * sizeof(bf16_t);
  bf16_t* Vt  = (bf16_t*)ws;  ws += H_ELEMS * sizeof(bf16_t);
  bf16_t* Om  = (bf16_t*)ws;  ws += X_ELEMS * sizeof(bf16_t);

  float* outY = (float*)d_out;
  float* attn = outY + X_ELEMS;   // attn output region: (B, H, L_Q, L_K) fp32

  // 1) fp32 -> bf16 conversions
  {
    int n4x = (int)(X_ELEMS / 4), n4w = (int)(W_ELEMS / 4);
    cvt_f32_bf16<<<(n4x + 255) / 256, 256, 0, stream>>>(q, xq, n4x);
    cvt_f32_bf16<<<(n4x + 255) / 256, 256, 0, stream>>>(k, xk, n4x);
    cvt_f32_bf16<<<(n4x + 255) / 256, 256, 0, stream>>>(v, xv, n4x);
    cvt_f32_bf16<<<(n4w + 255) / 256, 256, 0, stream>>>(w_q, wqb, n4w);
    cvt_f32_bf16<<<(n4w + 255) / 256, 256, 0, stream>>>(w_k, wkb, n4w);
    cvt_f32_bf16<<<(n4w + 255) / 256, 256, 0, stream>>>(w_v, wvb, n4w);
    cvt_f32_bf16<<<(n4w + 255) / 256, 256, 0, stream>>>(w_o, wob, n4w);
  }

  // 2) Q/K/V projections (WMMA bf16, async-LDS staged weights)
  dim3 gProj(L_Q / 128, DMODEL / 64, BATCH);
  proj_kernel<<<gProj, 256, 0, stream>>>(xq, wqb, b_q, Qb, 0);
  proj_kernel<<<gProj, 256, 0, stream>>>(xk, wkb, b_k, Kb, 0);
  proj_kernel<<<gProj, 256, 0, stream>>>(xv, wvb, b_v, Vt, 1);

  // 3) masked scaled scores (WMMA bf16 -> fp32)
  dim3 gS(L_Q / 128, L_K / 64, BATCH * NHEADS);
  scores_kernel<<<gS, 256, 0, stream>>>(Qb, Kb, mask, attn);

  // 4) softmax rows
  softmax_kernel<<<BATCH * NHEADS * L_Q, 256, 0, stream>>>(attn);

  // 5) attn @ V (WMMA bf16, async-LDS staged V^T)
  dim3 gAV(L_Q / 128, 1, BATCH * NHEADS);
  av_kernel<<<gAV, 256, 0, stream>>>(attn, Vt, Om);

  // 6) output projection -> fp32 d_out
  dim3 gO(L_Q / 128, DMODEL / 64, BATCH);
  outproj_kernel<<<gO, 256, 0, stream>>>(Om, wob, b_o, outY);
}
